// CausalSelfAttention_38689065402929
// MI455X (gfx1250) — matrix operand order repaired and checked by reading, not compiled, not run on hardware
//
#include <hip/hip_runtime.h>
#include <hip/hip_bf16.h>

// ---------------------------------------------------------------------------
// CDNA5 / gfx1250 causal self-attention (pair mask), bf16 WMMA pipeline.
//   B=4, T=2048, C=1024, H=16, hd=64
// ---------------------------------------------------------------------------

typedef __bf16 bf16;
typedef __attribute__((ext_vector_type(16))) __bf16 v16bf;
typedef __attribute__((ext_vector_type(8)))  float  v8f;

#define NB 4
#define NT 2048
#define NC 1024
#define NH 16
#define HD 64

__device__ __forceinline__ v8f wmma_bf16(v16bf a, v16bf b, v8f c) {
    // D = A(16x32) * B(32x16) + C, fp32 accum
    return __builtin_amdgcn_wmma_f32_16x16x32_bf16(
        /*neg_a=*/false, a, /*neg_b=*/false, b,
        /*c_mod=*/(short)0, c, /*reuse_a=*/false, /*reuse_b=*/false);
}

// A operand: tile stored row-major [16 M][K] (strideE elems, strideE%8==0).
// lane L: M=L%16, base=8*(L>=16); elems 0..7 = K base..base+7, 8..15 = K base+16..base+23
__device__ __forceinline__ v16bf opA(const bf16* tile, int lane, int strideE) {
    int m = lane & 15, hi = lane >> 4;
    const bf16* p = tile + m * strideE + hi * 8;
    union { v16bf v; uint4 u[2]; } r;
    r.u[0] = *(const uint4*)p;
    r.u[1] = *(const uint4*)(p + 16);
    return r.v;
}

// B operand: tile stored row-major [16 N][K] (strideE elems).
// lane L: N=L%16, base=8*(L>=16); elems 0..7 = K base..base+7, 8..15 = K base+16..base+23
__device__ __forceinline__ v16bf opB(const bf16* tile, int lane, int strideE) {
    int n = lane & 15, hi = lane >> 4;
    const bf16* p = tile + n * strideE + hi * 8;
    union { v16bf v; uint4 u[2]; } r;
    r.u[0] = *(const uint4*)p;
    r.u[1] = *(const uint4*)(p + 16);
    return r.v;
}

// ---------------------------------------------------------------------------
// Kernel 1: qkv = x @ w_attn^T + b_attn ; scatter q (pre-scaled), k, v^T as bf16
//   M=8192, N=3072, K=1024. Block tile 128x128, 8 waves (4x2 of 32x64).
//   Epilogue: stage 128x128 tile in LDS (transposed for v), coalesced b128 out.
// ---------------------------------------------------------------------------
#define GS 40    // LDS row stride (elems) for 32-wide K tiles (80B, 16B-aligned)
#define OS 136   // LDS row stride for 128-wide output staging tile

__global__ __launch_bounds__(256) void qkv_gemm_kernel(
    const float* __restrict__ x, const float* __restrict__ w,
    const float* __restrict__ bias,
    bf16* __restrict__ qb, bf16* __restrict__ kb, bf16* __restrict__ vtb)
{
    __shared__ __align__(16) bf16 smem[128 * OS];   // 34816 B; aliases A/B tiles
    bf16* ldsA = smem;
    bf16* ldsB = smem + 128 * GS;

    const int tid = threadIdx.x, lane = tid & 31, wv = tid >> 5;
    const int mBase = blockIdx.x * 128, nBase = blockIdx.y * 128;
    const int wr = wv >> 1, wc = wv & 1;   // wave sub-tile: rows wr*32, cols wc*64

    v8f acc[2][4];
    #pragma unroll
    for (int i = 0; i < 2; ++i)
        #pragma unroll
        for (int j = 0; j < 4; ++j)
            acc[i][j] = (v8f){0.f, 0.f, 0.f, 0.f, 0.f, 0.f, 0.f, 0.f};

    for (int k0 = 0; k0 < NC; k0 += 32) {
        // stage A (x) and B (w_attn) 128x32 fp32 -> bf16 tiles
        #pragma unroll
        for (int j = 0; j < 4; ++j) {
            int c = j * 256 + tid;            // 1024 chunks of 4 floats
            int row = c >> 3, col = (c & 7) * 4;
            float4 fa = *(const float4*)(x + (size_t)(mBase + row) * NC + k0 + col);
            bf16* da = ldsA + row * GS + col;
            da[0] = (bf16)fa.x; da[1] = (bf16)fa.y; da[2] = (bf16)fa.z; da[3] = (bf16)fa.w;
            float4 fb = *(const float4*)(w + (size_t)(nBase + row) * NC + k0 + col);
            bf16* db = ldsB + row * GS + col;
            db[0] = (bf16)fb.x; db[1] = (bf16)fb.y; db[2] = (bf16)fb.z; db[3] = (bf16)fb.w;
        }
        __syncthreads();

        const bf16* At = ldsA + (wr * 32) * GS;
        const bf16* Bt = ldsB + (wc * 64) * GS;
        v16bf a0 = opA(At, lane, GS);
        v16bf a1 = opA(At + 16 * GS, lane, GS);
        #pragma unroll
        for (int nt = 0; nt < 4; ++nt) {
            v16bf b = opB(Bt + nt * 16 * GS, lane, GS);
            acc[0][nt] = wmma_bf16(a0, b, acc[0][nt]);
            acc[1][nt] = wmma_bf16(a1, b, acc[1][nt]);
        }
        __syncthreads();
    }

    // ---- epilogue: block-uniform region; hoisted branch; LDS staging; b128 out
    const int region = nBase >> 10;        // 0=q, 1=k, 2=v  (tile never straddles)
    const int hi = lane >> 4, nl = lane & 15;

    float bsv[4];
    #pragma unroll
    for (int nt = 0; nt < 4; ++nt)
        bsv[nt] = bias[nBase + wc * 64 + nt * 16 + nl];

    if (region < 2) {
        const float sc = (region == 0) ? 0.125f : 1.0f;   // 1/sqrt(64) for q
        #pragma unroll
        for (int mt = 0; mt < 2; ++mt)
            #pragma unroll
            for (int nt = 0; nt < 4; ++nt)
                #pragma unroll
                for (int r = 0; r < 8; ++r) {
                    int t_loc = wr * 32 + mt * 16 + r + 8 * hi;
                    int n_loc = wc * 64 + nt * 16 + nl;
                    smem[t_loc * OS + n_loc] = (bf16)((acc[mt][nt][r] + bsv[nt]) * sc);
                }
    } else {
        #pragma unroll
        for (int mt = 0; mt < 2; ++mt)
            #pragma unroll
            for (int nt = 0; nt < 4; ++nt)
                #pragma unroll
                for (int r = 0; r < 8; ++r) {
                    int t_loc = wr * 32 + mt * 16 + r + 8 * hi;
                    int n_loc = wc * 64 + nt * 16 + nl;
                    smem[n_loc * OS + t_loc] = (bf16)(acc[mt][nt][r] + bsv[nt]); // v^T
                }
    }
    __syncthreads();

    const int bb = mBase >> 11, t0 = mBase & (NT - 1);
    if (region < 2) {
        bf16* dst = (region == 0) ? qb : kb;
        const int h0 = (nBase & (NC - 1)) >> 6;   // first head in this tile
        #pragma unroll
        for (int j = 0; j < 8; ++j) {
            int c = j * 256 + tid;                // 2048 chunks of 8 bf16
            int trow = c >> 4, ncol = (c & 15) * 8;
            size_t idx = ((size_t)(bb * NH + h0 + (ncol >> 6)) * NT + t0 + trow) * HD
                       + (ncol & 63);
            *(uint4*)(dst + idx) = *(const uint4*)(smem + trow * OS + ncol);
        }
    } else {
        const int rowBase = bb * NC + (nBase - 2 * NC);  // (b*16+h)*64+f flattened
        #pragma unroll
        for (int j = 0; j < 8; ++j) {
            int c = j * 256 + tid;
            int frow = c >> 4, tcol = (c & 15) * 8;
            *(uint4*)(vtb + (size_t)(rowBase + frow) * NT + t0 + tcol) =
                *(const uint4*)(smem + frow * OS + tcol);
        }
    }
}

// ---------------------------------------------------------------------------
// Kernel 2: flash attention with pair mask. Block = (b, h, 128-query tile),
// 8 waves * 16 queries. Streams 128-key chunks; 32-key softmax substeps.
// ---------------------------------------------------------------------------
#define QS 72   // [128 q][64 f] stride
#define KS 72   // [128 k][64 f] stride
#define VS 136  // [64 f][128 k] stride
#define PS 40   // per-wave [16 q][32 k] stride

__global__ __launch_bounds__(256) void attn_kernel(
    const bf16* __restrict__ qb, const bf16* __restrict__ kb,
    const bf16* __restrict__ vtb, bf16* __restrict__ yb)
{
    __shared__ __align__(16) bf16 qs[128 * QS];
    __shared__ __align__(16) bf16 ks[128 * KS];   // reused for output staging
    __shared__ __align__(16) bf16 vs[64 * VS];
    __shared__ __align__(16) bf16 ps[8 * 16 * PS];

    const int tid = threadIdx.x, lane = tid & 31, wv = tid >> 5;
    const int qt = blockIdx.x & 15;
    const int h  = (blockIdx.x >> 4) & 15;
    const int bb = blockIdx.x >> 8;
    const size_t bh = (size_t)bb * NH + h;
    const bf16* q  = qb  + bh * NT * HD;
    const bf16* k  = kb  + bh * NT * HD;
    const bf16* vt = vtb + bh * HD * NT;

    // stage the 128x64 query tile once
    #pragma unroll
    for (int j = 0; j < 4; ++j) {
        int c = j * 256 + tid;            // 1024 chunks of 8 bf16
        int row = c >> 3, col = (c & 7) * 8;
        *(uint4*)(qs + row * QS + col) =
            *(const uint4*)(q + (size_t)(qt * 128 + row) * HD + col);
    }
    __syncthreads();

    // hoist Q operands: invariant for the whole kernel
    const v16bf a0q = opA(qs + wv * 16 * QS, lane, QS);        // f 0..31
    const v16bf a1q = opA(qs + wv * 16 * QS + 32, lane, QS);   // f 32..63

    const float NEG = -__builtin_huge_valf();
    float mi[8], li[8];
    v8f o[4];
    #pragma unroll
    for (int r = 0; r < 8; ++r) { mi[r] = NEG; li[r] = 0.f; }
    #pragma unroll
    for (int nt = 0; nt < 4; ++nt)
        o[nt] = (v8f){0.f, 0.f, 0.f, 0.f, 0.f, 0.f, 0.f, 0.f};

    const int hi = lane >> 4, nl = lane & 15;
    const int qrow0 = qt * 128 + wv * 16;
    bf16* psw = ps + wv * 16 * PS;

    const int nChunks = qt + 1;               // causal: keys <= last query of tile
    for (int kc = 0; kc < nChunks; ++kc) {
        // stage 128 keys x 64 features, and v^T 64 features x 128 keys
        #pragma unroll
        for (int j = 0; j < 4; ++j) {
            int c = j * 256 + tid;
            int krow = c >> 3, kcol = (c & 7) * 8;
            *(uint4*)(ks + krow * KS + kcol) =
                *(const uint4*)(k + (size_t)(kc * 128 + krow) * HD + kcol);
            int vrow = c >> 4, vcol = (c & 15) * 8;
            *(uint4*)(vs + vrow * VS + vcol) =
                *(const uint4*)(vt + (size_t)vrow * NT + kc * 128 + vcol);
        }
        __syncthreads();

        for (int js = 0; js < 4; ++js) {      // 32 keys per substep
            v8f s0 = (v8f){0.f, 0.f, 0.f, 0.f, 0.f, 0.f, 0.f, 0.f};
            v8f s1 = s0;
            {
                const bf16* kt0 = ks + (js * 32) * KS;
                s0 = wmma_bf16(a0q, opB(kt0, lane, KS), s0);
                s0 = wmma_bf16(a1q, opB(kt0 + 32, lane, KS), s0);
                const bf16* kt1 = ks + (js * 32 + 16) * KS;
                s1 = wmma_bf16(a0q, opB(kt1, lane, KS), s1);
                s1 = wmma_bf16(a1q, opB(kt1 + 32, lane, KS), s1);
            }
            const int key0 = kc * 128 + js * 32 + nl;
            #pragma unroll
            for (int r = 0; r < 8; ++r) {
                int qi = qrow0 + r + 8 * hi;
                int k0i = key0, k1i = key0 + 16;
                float v0 = s0[r], v1 = s1[r];
                // pair mask: causal, and odd qi can't attend to qi-1
                bool ok0 = (k0i <= qi) && !((qi & 1) && (k0i == qi - 1));
                bool ok1 = (k1i <= qi) && !((qi & 1) && (k1i == qi - 1));
                v0 = ok0 ? v0 : NEG;
                v1 = ok1 ? v1 : NEG;
                float rowm = fmaxf(v0, v1);
                rowm = fmaxf(rowm, __shfl_xor(rowm, 1));
                rowm = fmaxf(rowm, __shfl_xor(rowm, 2));
                rowm = fmaxf(rowm, __shfl_xor(rowm, 4));
                rowm = fmaxf(rowm, __shfl_xor(rowm, 8));
                float mnew = fmaxf(mi[r], rowm);
                float fr = __expf(mi[r] - mnew);
                float p0 = __expf(v0 - mnew);
                float p1 = __expf(v1 - mnew);
                float psum = p0 + p1;
                psum += __shfl_xor(psum, 1);
                psum += __shfl_xor(psum, 2);
                psum += __shfl_xor(psum, 4);
                psum += __shfl_xor(psum, 8);
                li[r] = li[r] * fr + psum;
                mi[r] = mnew;
                #pragma unroll
                for (int nt = 0; nt < 4; ++nt) o[nt][r] *= fr;
                // re-layout probs: D layout -> row-major [16 q][32 k] in LDS
                psw[(r + 8 * hi) * PS + nl]      = (bf16)p0;
                psw[(r + 8 * hi) * PS + 16 + nl] = (bf16)p1;
            }
            asm volatile("s_wait_dscnt 0" ::: "memory");  // same-wave LDS RAW
            v16bf ap = opA(psw, lane, PS);                // P as A operand (16x32)
            #pragma unroll
            for (int nt = 0; nt < 4; ++nt) {
                v16bf bv = opB(vs + (nt * 16) * VS + js * 32, lane, VS);
                o[nt] = wmma_bf16(ap, bv, o[nt]);
            }
        }
        __syncthreads();
    }

    // finalize: o /= l; stage 128x64 tile in LDS (reuse ks), coalesced b128 out
    #pragma unroll
    for (int r = 0; r < 8; ++r) {
        float inv = 1.0f / li[r];
        int t_loc = wv * 16 + r + 8 * hi;
        #pragma unroll
        for (int nt = 0; nt < 4; ++nt)
            ks[t_loc * KS + nt * 16 + nl] = (bf16)(o[nt][r] * inv);
    }
    __syncthreads();
    #pragma unroll
    for (int j = 0; j < 4; ++j) {
        int c = j * 256 + tid;               // 1024 chunks of 8 bf16
        int trow = c >> 3, fcol = (c & 7) * 8;
        *(uint4*)(yb + ((size_t)bb * NT + qt * 128 + trow) * NC + h * HD + fcol) =
            *(const uint4*)(ks + trow * KS + fcol);
    }
}

// ---------------------------------------------------------------------------
// Kernel 3: out = y @ w_proj^T + b_proj (fp32 out). M=8192, N=K=1024.
// ---------------------------------------------------------------------------
__global__ __launch_bounds__(256) void proj_gemm_kernel(
    const bf16* __restrict__ y, const float* __restrict__ w,
    const float* __restrict__ bias, float* __restrict__ out)
{
    __shared__ __align__(16) bf16 ldsA[128 * GS];
    __shared__ __align__(16) bf16 ldsB[128 * GS];
    const int tid = threadIdx.x, lane = tid & 31, wv = tid >> 5;
    const int mBase = blockIdx.x * 128, nBase = blockIdx.y * 128;
    const int wr = wv >> 1, wc = wv & 1;

    v8f acc[2][4];
    #pragma unroll
    for (int i = 0; i < 2; ++i)
        #pragma unroll
        for (int j = 0; j < 4; ++j)
            acc[i][j] = (v8f){0.f, 0.f, 0.f, 0.f, 0.f, 0.f, 0.f, 0.f};

    for (int k0 = 0; k0 < NC; k0 += 32) {
        #pragma unroll
        for (int j = 0; j < 2; ++j) {        // A: 512 chunks of 8 bf16
            int c = j * 256 + tid;
            int row = c >> 2, col = (c & 3) * 8;
            *(uint4*)(ldsA + row * GS + col) =
                *(const uint4*)(y + (size_t)(mBase + row) * NC + k0 + col);
        }
        #pragma unroll
        for (int j = 0; j < 4; ++j) {        // B: fp32 -> bf16
            int c = j * 256 + tid;
            int row = c >> 3, col = (c & 7) * 4;
            float4 fb = *(const float4*)(w + (size_t)(nBase + row) * NC + k0 + col);
            bf16* db = ldsB + row * GS + col;
            db[0] = (bf16)fb.x; db[1] = (bf16)fb.y; db[2] = (bf16)fb.z; db[3] = (bf16)fb.w;
        }
        __syncthreads();

        const bf16* At = ldsA + (wr * 32) * GS;
        const bf16* Bt = ldsB + (wc * 64) * GS;
        v16bf a0 = opA(At, lane, GS);
        v16bf a1 = opA(At + 16 * GS, lane, GS);
        #pragma unroll
        for (int nt = 0; nt < 4; ++nt) {
            v16bf b = opB(Bt + nt * 16 * GS, lane, GS);
            acc[0][nt] = wmma_bf16(a0, b, acc[0][nt]);
            acc[1][nt] = wmma_bf16(a1, b, acc[1][nt]);
        }
        __syncthreads();
    }

    const int hi = lane >> 4, nl = lane & 15;
    #pragma unroll
    for (int mt = 0; mt < 2; ++mt)
        #pragma unroll
        for (int nt = 0; nt < 4; ++nt)
            #pragma unroll
            for (int r = 0; r < 8; ++r) {
                int m = mBase + wr * 32 + mt * 16 + r + 8 * hi;
                int n = nBase + wc * 64 + nt * 16 + nl;
                out[(size_t)m * NC + n] = acc[mt][nt][r] + bias[n];
            }
}

// ---------------------------------------------------------------------------
extern "C" void kernel_launch(void* const* d_in, const int* in_sizes, int n_in,
                              void* d_out, int out_size, void* d_ws, size_t ws_size,
                              hipStream_t stream) {
    const float* x      = (const float*)d_in[0];
    const float* w_attn = (const float*)d_in[1];
    const float* b_attn = (const float*)d_in[2];
    const float* w_proj = (const float*)d_in[3];
    const float* b_proj = (const float*)d_in[4];
    float* out = (float*)d_out;

    char* ws = (char*)d_ws;
    const size_t SEG = (size_t)NB * NH * NT * HD * sizeof(bf16);  // 16 MB each
    bf16* qb  = (bf16*)(ws);
    bf16* kb  = (bf16*)(ws + SEG);
    bf16* vtb = (bf16*)(ws + 2 * SEG);
    bf16* yb  = (bf16*)(ws + 3 * SEG);

    dim3 gA((NB * NT) / 128, (3 * NC) / 128);   // 64 x 24
    qkv_gemm_kernel<<<gA, 256, 0, stream>>>(x, w_attn, b_attn, qb, kb, vtb);

    dim3 gB(NB * NH * (NT / 128));               // 1024
    attn_kernel<<<gB, 256, 0, stream>>>(qb, kb, vtb, yb);

    dim3 gC((NB * NT) / 128, NC / 128);          // 64 x 8
    proj_gemm_kernel<<<gC, 256, 0, stream>>>(yb, w_proj, b_proj, out);
}
